// ShiftedWindowMA_77945066488324
// MI455X (gfx1250) — compile-verified, matrix operand order checked by reading, not run on hardware
//
#include <hip/hip_runtime.h>

typedef __attribute__((ext_vector_type(16))) _Float16 v16h;
typedef __attribute__((ext_vector_type(8)))  _Float16 v8h;
typedef __attribute__((ext_vector_type(4)))  _Float16 v4h;
typedef __attribute__((ext_vector_type(8)))  float    v8f;
typedef __attribute__((ext_vector_type(4)))  float    v4f;

#define NH_ 4
#define WTOK 64                    // tokens padded 49 -> 64
#define ECH 32                     // channels per head
#define WIN_ELT (WTOK * ECH)       // 2048 halves per (window, head)
#define NWIN 256                   // 16x16 windows per (b, head)
#define HEADSTRIDE (NWIN * WIN_ELT)// 524288 halves

// Load one 16-bit A/B WMMA fragment (16x32, K-split across the two lane
// halves) from a row-major f16 matrix. p0 points at [row][k_half_base].
__device__ inline v16h frag_ld(const _Float16* p0) {
    union { v16h v; v8h h[2]; } u;
    u.h[0] = *(const v8h*)(p0);
    u.h[1] = *(const v8h*)(p0 + 16);
    return u.v;
}

// ---------------------------------------------------------------- Wqk -> f16
__global__ __launch_bounds__(256)
void wcvt_kernel(const float* __restrict__ w, _Float16* __restrict__ wf) {
    int g = blockIdx.x * 256 + threadIdx.x;      // 8192 groups of 4 (exact)
    v4f d = *(const v4f*)(w + (size_t)g * 4);
    v4h h; h.x = (_Float16)d.x; h.y = (_Float16)d.y;
           h.z = (_Float16)d.z; h.w = (_Float16)d.w;
    *(v4h*)(wf + (size_t)g * 4) = h;
}

// --------------------------------------------- QK projection GEMM + scatter
// M = 200704 pixels, K = 128, N = 256.  128-pixel tile per block, 8 waves,
// each wave owns 16 pixel rows and loops over 16 N-tiles.
__global__ __launch_bounds__(256)
void proj_kernel(const float* __restrict__ x, const float* __restrict__ bqk,
                 const _Float16* __restrict__ Wf,
                 _Float16* __restrict__ qb, _Float16* __restrict__ kb) {
    __shared__ _Float16 xt[128 * 128];
    const int tid = threadIdx.x;
    const int P0  = blockIdx.x * 128;

    // stage x tile as f16 into LDS (coalesced b128 loads, ds_store_b64)
    const float* xg = x + (size_t)P0 * 128;
    #pragma unroll
    for (int it = 0; it < 16; ++it) {
        int f = tid * 4 + it * 1024;
        v4f d = *(const v4f*)(xg + f);
        v4h h; h.x = (_Float16)d.x; h.y = (_Float16)d.y;
               h.z = (_Float16)d.z; h.w = (_Float16)d.w;
        *(v4h*)(xt + f) = h;
    }
    __syncthreads();

    const int wave  = tid >> 5, lane = tid & 31;
    const int laneN = lane & 15, hi = lane >> 4;
    const int ko    = hi * 8;
    const int m0    = wave * 16;

    v16h a[4];
    #pragma unroll
    for (int kt = 0; kt < 4; ++kt)
        a[kt] = frag_ld(xt + (m0 + laneN) * 128 + kt * 32 + ko);

    // per-row scatter base: roll(-4,-4) + window partition
    int wb[8];
    #pragma unroll
    for (int r = 0; r < 8; ++r) {
        int pix = P0 + m0 + r + hi * 8;
        int b   = pix / 12544;
        int rem = pix - b * 12544;
        int ys  = rem / 112;
        int xs  = rem - ys * 112;
        int yr  = ys - 4; if (yr < 0) yr += 112;
        int xr  = xs - 4; if (xr < 0) xr += 112;
        int wx  = yr / 7, iy = yr - wx * 7;
        int wy  = xr / 7, ix = xr - wy * 7;
        int t   = iy * 7 + ix;
        wb[r] = ((b * NH_) * NWIN + wx * 16 + wy) * WIN_ELT + t * ECH;
    }

    #pragma unroll
    for (int nt = 0; nt < 16; ++nt) {
        int n = nt * 16 + laneN;                 // output channel (column)
        float bv = bqk[n];
        v8f c;
        #pragma unroll
        for (int i = 0; i < 8; ++i) c[i] = bv;   // bias folded into C
        #pragma unroll
        for (int kt = 0; kt < 4; ++kt) {
            v16h bf = frag_ld(Wf + n * 128 + kt * 32 + ko);  // B^T rows = Wqk rows
            c = __builtin_amdgcn_wmma_f32_16x16x32_f16(
                    false, a[kt], false, bf, (short)0, c, false, false);
        }
        int cch  = n >> 1;
        int e    = cch >> 2;
        int head = cch & 3;
        _Float16* dst = (n & 1) ? kb : qb;
        float scale = (n & 1) ? 1.0f : 0.1767766953f;   // fold 1/sqrt(32) into q
        #pragma unroll
        for (int r = 0; r < 8; ++r)
            dst[wb[r] + head * HEADSTRIDE + e] = (_Float16)(c[r] * scale);
    }
}

// ------------------------------------------------- v -> f16, transposed tile
// vT layout per (window, head): [ch 0..31][token 0..63]
__global__ __launch_bounds__(256)
void vscatter_kernel(const float* __restrict__ v, _Float16* __restrict__ vb) {
    int g = blockIdx.x * 256 + threadIdx.x;
    if (g >= 6422528) return;                    // 16*112*112*128/4
    int f4  = g * 4;
    int pix = f4 >> 7;
    int c0  = f4 & 127;                          // multiple of 4 -> heads 0..3
    int b   = pix / 12544;
    int rem = pix - b * 12544;
    int ys  = rem / 112;
    int xs  = rem - ys * 112;
    int yr  = ys - 4; if (yr < 0) yr += 112;
    int xr  = xs - 4; if (xr < 0) xr += 112;
    int wx  = yr / 7, iy = yr - wx * 7;
    int wy  = xr / 7, ix = xr - wy * 7;
    int t   = iy * 7 + ix;
    int e   = c0 >> 2;
    v4f d = *(const v4f*)(v + (size_t)f4);
    size_t base = ((size_t)(b * NH_) * NWIN + wx * 16 + wy) * WIN_ELT + e * WTOK + t;
    vb[base + 0 * (size_t)HEADSTRIDE] = (_Float16)d.x;
    vb[base + 1 * (size_t)HEADSTRIDE] = (_Float16)d.y;
    vb[base + 2 * (size_t)HEADSTRIDE] = (_Float16)d.z;
    vb[base + 3 * (size_t)HEADSTRIDE] = (_Float16)d.w;
}

// ------------------------------------------------------- windowed attention
// One block per (b, wx, wy); one wave per head.  q/k/v bulk-staged into LDS
// with global_load_async_to_lds_b128 (ASYNCcnt), 64x64 scores via WMMA,
// register softmax, P reuses the q+k LDS region of the same head (its q/k
// are already in VGPRs; LDS ops are in-order within a wave), P@V via WMMA,
// fused window-merge + roll(+3).
__global__ __launch_bounds__(128)
void attn_kernel(const _Float16* __restrict__ qb, const _Float16* __restrict__ kb,
                 const _Float16* __restrict__ vb, const float* __restrict__ pos,
                 float* __restrict__ out) {
    __shared__ _Float16 sQK[4][4096];   // [head]: [0..2047]=q, [2048..4095]=k -> later P[64][64]
    __shared__ _Float16 sV[4][2048];    // [head]: vT [ch][token]
    __shared__ float pe[169];

    const int tid  = threadIdx.x;
    const int head = tid >> 5, lane = tid & 31;
    const int laneN = lane & 15, hi = lane >> 4, ko = hi * 8;
    const int blk = blockIdx.x;
    const int b = blk >> 8, wx = (blk >> 4) & 15, wy = blk & 15;
    const size_t base = ((size_t)(b * NH_ + head) * NWIN + wx * 16 + wy) * WIN_ELT;

    // ---- async bulk stage: 4KB q + 4KB k + 4KB v per wave, 512B per issue
    {
        unsigned lq = (unsigned)(size_t)(const void*)&sQK[head][0]    + lane * 16u;
        unsigned lk = (unsigned)(size_t)(const void*)&sQK[head][2048] + lane * 16u;
        unsigned lv = (unsigned)(size_t)(const void*)&sV[head][0]     + lane * 16u;
        unsigned long long gq = (unsigned long long)(const void*)(qb + base) + lane * 16u;
        unsigned long long gk = (unsigned long long)(const void*)(kb + base) + lane * 16u;
        unsigned long long gv = (unsigned long long)(const void*)(vb + base) + lane * 16u;
        #pragma unroll
        for (int i = 0; i < 8; ++i) {
            unsigned off = i * 512u;
            asm volatile("global_load_async_to_lds_b128 %0, %1, off"
                         :: "v"(lq + off), "v"(gq + (unsigned long long)off) : "memory");
            asm volatile("global_load_async_to_lds_b128 %0, %1, off"
                         :: "v"(lk + off), "v"(gk + (unsigned long long)off) : "memory");
            asm volatile("global_load_async_to_lds_b128 %0, %1, off"
                         :: "v"(lv + off), "v"(gv + (unsigned long long)off) : "memory");
        }
    }
    for (int i = tid; i < 169; i += 128) pe[i] = pos[i];
    asm volatile("s_wait_asynccnt 0x0" ::: "memory");
    __syncthreads();

    const _Float16* q  = &sQK[head][0];
    const _Float16* k  = &sQK[head][2048];
    const _Float16* vp = &sV[head][0];

    // -------- S = q @ k^T  (M=64, N=64, K=32)
    v16h aq[4];
    #pragma unroll
    for (int mi = 0; mi < 4; ++mi)
        aq[mi] = frag_ld(q + (mi * 16 + laneN) * 32 + ko);

    v8f acc[4][4];
    #pragma unroll
    for (int mi = 0; mi < 4; ++mi)
        #pragma unroll
        for (int ni = 0; ni < 4; ++ni)
            #pragma unroll
            for (int i = 0; i < 8; ++i) acc[mi][ni][i] = 0.0f;

    #pragma unroll
    for (int ni = 0; ni < 4; ++ni) {
        v16h bk = frag_ld(k + (ni * 16 + laneN) * 32 + ko);
        #pragma unroll
        for (int mi = 0; mi < 4; ++mi)
            acc[mi][ni] = __builtin_amdgcn_wmma_f32_16x16x32_f16(
                false, aq[mi], false, bk, (short)0, acc[mi][ni], false, false);
    }

    // -------- bias + shift masks + softmax (rows live in 16-lane groups)
    const bool lastX = (wx == 15), lastY = (wy == 15);
    _Float16* P = &sQK[head][0];                 // reuse q+k region for P[64][64]
    #pragma unroll
    for (int mi = 0; mi < 4; ++mi) {
        #pragma unroll
        for (int r = 0; r < 8; ++r) {
            int row = mi * 16 + r + hi * 8;
            int rc  = row < 49 ? row : 48;
            int ir  = rc / 7, jr = rc - ir * 7;
            float sv[4];
            float mx = -3.0e38f;
            #pragma unroll
            for (int ni = 0; ni < 4; ++ni) {
                int col = ni * 16 + laneN;
                int cc  = col < 49 ? col : 48;
                int ic  = cc / 7, jc = cc - ic * 7;
                float s = acc[mi][ni][r] + pe[(ic - ir + 6) * 13 + (jc - jr + 6)];
                bool dead = (col >= 49)
                         || (lastX && ((ir >= 4) != (ic >= 4)))
                         || (lastY && ((jr >= 4) != (jc >= 4)));
                s = dead ? -3.0e38f : s;
                sv[ni] = s;
                mx = fmaxf(mx, s);
            }
            #pragma unroll
            for (int d = 1; d < 16; d <<= 1) mx = fmaxf(mx, __shfl_xor(mx, d, 32));
            float sum = 0.0f;
            #pragma unroll
            for (int ni = 0; ni < 4; ++ni) { sv[ni] = __expf(sv[ni] - mx); sum += sv[ni]; }
            #pragma unroll
            for (int d = 1; d < 16; d <<= 1) sum += __shfl_xor(sum, d, 32);
            float inv = 1.0f / sum;
            #pragma unroll
            for (int ni = 0; ni < 4; ++ni)
                P[row * 64 + ni * 16 + laneN] = (_Float16)(sv[ni] * inv);
        }
    }

    // -------- O = P @ V  (M=64, N=32, K=64)
    v8f o[4][2];
    #pragma unroll
    for (int mi = 0; mi < 4; ++mi)
        #pragma unroll
        for (int ni = 0; ni < 2; ++ni)
            #pragma unroll
            for (int i = 0; i < 8; ++i) o[mi][ni][i] = 0.0f;

    #pragma unroll
    for (int kt = 0; kt < 2; ++kt) {
        v16h ap[4];
        #pragma unroll
        for (int mi = 0; mi < 4; ++mi)
            ap[mi] = frag_ld(P + (mi * 16 + laneN) * 64 + kt * 32 + ko);
        #pragma unroll
        for (int ni = 0; ni < 2; ++ni) {
            v16h bv = frag_ld(vp + (ni * 16 + laneN) * 64 + kt * 32 + ko);
            #pragma unroll
            for (int mi = 0; mi < 4; ++mi)
                o[mi][ni] = __builtin_amdgcn_wmma_f32_16x16x32_f16(
                    false, ap[mi], false, bv, (short)0, o[mi][ni], false, false);
        }
    }

    // -------- window merge + roll(+3,+3), channel = e*4 + head
    #pragma unroll
    for (int mi = 0; mi < 4; ++mi) {
        #pragma unroll
        for (int r = 0; r < 8; ++r) {
            int row = mi * 16 + r + hi * 8;
            if (row < 49) {
                int i = row / 7, j = row - i * 7;
                int y  = wx * 7 + i + 3; if (y  >= 112) y  -= 112;
                int xo = wy * 7 + j + 3; if (xo >= 112) xo -= 112;
                size_t po = (((size_t)b * 112 + y) * 112 + xo) * 128;
                #pragma unroll
                for (int ni = 0; ni < 2; ++ni) {
                    int ch = (ni * 16 + laneN) * 4 + head;
                    out[po + ch] = o[mi][ni][r];
                }
            }
        }
    }
}

extern "C" void kernel_launch(void* const* d_in, const int* in_sizes, int n_in,
                              void* d_out, int out_size, void* d_ws, size_t ws_size,
                              hipStream_t stream) {
    (void)in_sizes; (void)n_in; (void)out_size; (void)ws_size;
    const float* x   = (const float*)d_in[0];
    const float* v   = (const float*)d_in[1];
    const float* Wqk = (const float*)d_in[2];
    const float* bqk = (const float*)d_in[3];
    const float* pos = (const float*)d_in[4];
    float* out = (float*)d_out;

    char* ws = (char*)d_ws;
    _Float16* Wf = (_Float16*)ws;                       // 64 KB
    _Float16* qb = (_Float16*)(ws + 65536);             // 64 MiB
    _Float16* kb = qb + (size_t)33554432;               // 64 MiB
    _Float16* vb = kb + (size_t)33554432;               // 64 MiB

    wcvt_kernel    <<<dim3(32),    dim3(256), 0, stream>>>(Wqk, Wf);
    proj_kernel    <<<dim3(1568),  dim3(256), 0, stream>>>(x, bqk, Wf, qb, kb);
    vscatter_kernel<<<dim3(25088), dim3(256), 0, stream>>>(v, vb);
    attn_kernel    <<<dim3(4096),  dim3(128), 0, stream>>>(qb, kb, vb, pos, out);
}
